// Encoder1DBlock_83717502533915
// MI455X (gfx1250) — compile-verified, matrix operand order checked by reading, not run on hardware
//
#include <hip/hip_runtime.h>

// ---------------------------------------------------------------------------
// Encoder block (pre-LN, relative attention, MLP) for gfx1250 / MI455X.
// All matmuls via v_wmma_f32_16x16x32_bf16 (fp32 accumulate).
// Global->LDS staging via GLOBAL_LOAD_ASYNC_TO_LDS_B128 (ASYNCcnt) with
// double buffering in the GEMM K-loop.
// B=4 S=1024 D=1024 H=16 DH=64 MLP=4096 R=16
// ---------------------------------------------------------------------------

typedef __attribute__((ext_vector_type(16))) __bf16          v16bf;
typedef __attribute__((ext_vector_type(16))) unsigned short  v16u;
typedef __attribute__((ext_vector_type(8)))  float           v8f;
typedef __attribute__((ext_vector_type(4)))  int             v4i;

#define GAS __attribute__((address_space(1)))
#define LAS __attribute__((address_space(3)))

#if defined(__has_builtin)
#  if __has_builtin(__builtin_amdgcn_global_load_async_to_lds_b128)
#    define HAVE_ASYNC_LDS 1
#  else
#    define HAVE_ASYNC_LDS 0
#  endif
#else
#  define HAVE_ASYNC_LDS 0
#endif

static __device__ __forceinline__ unsigned short f2bf(float f) {
    union { float f; unsigned u; } v; v.f = f;
    unsigned r = v.u + 0x7FFFu + ((v.u >> 16) & 1u);   // round-to-nearest-even
    return (unsigned short)(r >> 16);
}
static __device__ __forceinline__ float bf2f(unsigned short u) {
    union { unsigned u; float f; } v; v.u = ((unsigned)u) << 16;
    return v.f;
}

// 16B global -> LDS copy; async (ASYNCcnt-tracked) when the toolchain has it.
static __device__ __forceinline__ void cp_b128(const unsigned short* g, unsigned short* l) {
#if HAVE_ASYNC_LDS
    GAS const unsigned short* g1 = (GAS const unsigned short*)g;  // flat -> global AS
    LAS unsigned short*       l3 = (LAS unsigned short*)l;        // flat -> LDS AS
    __builtin_amdgcn_global_load_async_to_lds_b128((GAS v4i*)g1, (LAS v4i*)l3, 0, 0);
#else
    *(uint4*)l = *(const uint4*)g;
#endif
}
static __device__ __forceinline__ void wait_cp() {
#if HAVE_ASYNC_LDS
#  if __has_builtin(__builtin_amdgcn_s_wait_asynccnt)
    __builtin_amdgcn_s_wait_asynccnt(0);
#  else
    asm volatile("s_wait_asynccnt 0" ::: "memory");
#  endif
#endif
}

// Build a 16x32 bf16 WMMA operand fragment from an LDS row of contiguous
// bf16 values (row-major, this lane's row). Element e -> K = (e>>3)*16 +
// half*8 + (e&7)  (CDNA5 16-bit A-operand layout; B operand loaded with the
// mirrored convention from a column-contiguous buffer).
static __device__ __forceinline__ v16bf ldfrag(const unsigned short* base, int half) {
    union { v16u v; uint4 q[2]; } u;
    u.q[0] = *(const uint4*)(base + half * 8);
    u.q[1] = *(const uint4*)(base + 16 + half * 8);
    return __builtin_bit_cast(v16bf, u.v);
}

static __device__ __forceinline__ v8f wmma_bf16(v16bf a, v16bf b, v8f c) {
    return __builtin_amdgcn_wmma_f32_16x16x32_bf16(false, a, false, b,
                                                   (short)0, c, false, false);
}

// ---------------------------------------------------------------------------
// f32 -> bf16 conversion (weights)
// ---------------------------------------------------------------------------
__global__ __launch_bounds__(256) void k_f32_to_bf16(const float* __restrict__ in,
                                                     unsigned short* __restrict__ out,
                                                     int n) {
    int i = blockIdx.x * 256 + threadIdx.x;
    if (i < n) out[i] = f2bf(in[i]);
}

// ---------------------------------------------------------------------------
// LayerNorm (row per block) -> bf16 output
// ---------------------------------------------------------------------------
__global__ __launch_bounds__(256) void k_layernorm_bf16(const float* __restrict__ x,
                                                        const float* __restrict__ scale,
                                                        const float* __restrict__ bias,
                                                        unsigned short* __restrict__ out,
                                                        int D) {
    const int row = blockIdx.x;
    const float* xr = x + (size_t)row * D;
    float s = 0.f, ss = 0.f;
    for (int i = threadIdx.x; i < D; i += 256) {
        float v = xr[i];
        s += v; ss += v * v;
    }
#pragma unroll
    for (int o = 16; o > 0; o >>= 1) {
        s  += __shfl_down(s,  o, 32);
        ss += __shfl_down(ss, o, 32);
    }
    __shared__ float rs[8], rss[8];
    int w = threadIdx.x >> 5, l = threadIdx.x & 31;
    if (l == 0) { rs[w] = s; rss[w] = ss; }
    __syncthreads();
    if (threadIdx.x == 0) {
        float a = 0.f, b = 0.f;
#pragma unroll
        for (int i = 0; i < 8; ++i) { a += rs[i]; b += rss[i]; }
        rs[0] = a; rss[0] = b;
    }
    __syncthreads();
    float mean = rs[0] / D;
    float var  = rss[0] / D - mean * mean;
    float inv  = rsqrtf(var + 1e-6f);
    for (int i = threadIdx.x; i < D; i += 256) {
        float v = (xr[i] - mean) * inv * scale[i] + bias[i];
        out[(size_t)row * D + i] = f2bf(v);
    }
}

// ---------------------------------------------------------------------------
// Generic WMMA GEMM: C[M,N] = A[M,K](bf16) @ B[K,N](bf16) + bias (+residual)
// Block tile 128x64, BK=32; 8 waves, each wave a 32x32 tile (2x2 WMMA accs).
// Double-buffered LDS; A tile staged with async global->LDS copies.
// ---------------------------------------------------------------------------
__global__ __launch_bounds__(256) void k_gemm_bf16(const unsigned short* __restrict__ A,
                                                   const unsigned short* __restrict__ Bm,
                                                   const float* __restrict__ bias,
                                                   const float* __restrict__ residual,
                                                   void* __restrict__ Cout,
                                                   int M, int N, int K,
                                                   int outBf16, int relu) {
    __shared__ unsigned short As[2][128 * 40];   // padded stride 40 (80B, 16B aligned)
    __shared__ unsigned short Bs[2][64 * 40];    // transposed [n][k]

    const int tid  = threadIdx.x;
    const int lane = tid & 31, half = lane >> 4, l16 = lane & 15;
    const int w = tid >> 5, wr = w >> 1, wc = w & 1;
    const int m0 = blockIdx.x * 128, n0 = blockIdx.y * 64;

    const int arow = tid >> 1, ak = (tid & 1) * 16;   // A loader: 128 rows x 2 halves
    const int brow = tid >> 3, bn = (tid & 7) * 8;    // B loader: 32 rows x 8 col-octets

    v8f acc[2][2] = {};

    auto stageA = [&](int k0, int buf) {
        const unsigned short* src = A + (size_t)(m0 + arow) * K + k0 + ak;
        cp_b128(src,     &As[buf][arow * 40 + ak]);
        cp_b128(src + 8, &As[buf][arow * 40 + ak + 8]);
    };
    auto stageB = [&](int k0, int buf) {
        const unsigned short* src = Bm + (size_t)(k0 + brow) * N + n0 + bn;
        uint4 v = *(const uint4*)src;
        if (k0 + 32 < K) __builtin_prefetch(src + (size_t)32 * N, 0, 1);  // global_prefetch
        unsigned short tmp[8];
        *(uint4*)tmp = v;
#pragma unroll
        for (int i = 0; i < 8; ++i) Bs[buf][(bn + i) * 40 + brow] = tmp[i];
    };

    stageA(0, 0);
    stageB(0, 0);
    wait_cp();
    __syncthreads();

    int buf = 0;
    for (int k0 = 0; k0 < K; k0 += 32) {
        // kick off next tile's copies while WMMAs run on the current tile
        if (k0 + 32 < K) {
            stageA(k0 + 32, buf ^ 1);
            stageB(k0 + 32, buf ^ 1);
        }

        v16bf af[2], bf[2];
#pragma unroll
        for (int i = 0; i < 2; ++i)
            af[i] = ldfrag(&As[buf][(wr * 32 + i * 16 + l16) * 40], half);
#pragma unroll
        for (int j = 0; j < 2; ++j)
            bf[j] = ldfrag(&Bs[buf][(wc * 32 + j * 16 + l16) * 40], half);

#pragma unroll
        for (int i = 0; i < 2; ++i)
#pragma unroll
            for (int j = 0; j < 2; ++j)
                acc[i][j] = wmma_bf16(af[i], bf[j], acc[i][j]);

        wait_cp();
        __syncthreads();
        buf ^= 1;
    }

    // epilogue: C layout per lane -> row = r + 8*half, col = l16 within tile
#pragma unroll
    for (int i = 0; i < 2; ++i)
#pragma unroll
        for (int j = 0; j < 2; ++j) {
            int col = n0 + wc * 32 + j * 16 + l16;
            float bv = bias ? bias[col] : 0.f;
#pragma unroll
            for (int r = 0; r < 8; ++r) {
                int row = m0 + wr * 32 + i * 16 + r + 8 * half;
                float v = acc[i][j][r] + bv;
                if (relu) v = fmaxf(v, 0.f);
                size_t idx = (size_t)row * N + col;
                if (residual) v += residual[idx];
                if (outBf16) ((unsigned short*)Cout)[idx] = f2bf(v);
                else         ((float*)Cout)[idx] = v;
            }
        }
}

// ---------------------------------------------------------------------------
// qr[row, j] = dot(q_bf[row, 0:64], rel_emb[j, 0:64])   row in [B*S*H), j<33
// (collapses the O(S^2*DH) relative einsum into an O(S*33*DH) table + gather)
// ---------------------------------------------------------------------------
__global__ __launch_bounds__(256) void k_qr(const unsigned short* __restrict__ qbf,
                                            const float* __restrict__ rel,
                                            float* __restrict__ qr, int rows) {
    int t = blockIdx.x * 256 + threadIdx.x;
    int row = t / 33, j = t % 33;
    if (row >= rows) return;
    const unsigned short* qp = qbf + (size_t)row * 64;
    const float* rp = rel + (size_t)j * 64;
    float s = 0.f;
#pragma unroll
    for (int d = 0; d < 64; ++d) s += bf2f(qp[d]) * rp[d];
    qr[(size_t)row * 36 + j] = s;
}

// ---------------------------------------------------------------------------
// Flash-style relative attention. Block = 256 thr (8 waves) handles 64 q-rows
// of one (b,h). Streams 64-wide k-tiles: S = Q@K^T via WMMA, add qr-gather +
// rel bias, scale, online softmax (P written as bf16), O += P@V via WMMA.
// ---------------------------------------------------------------------------
#define SCC 68
__global__ __launch_bounds__(256) void k_attention(const unsigned short* __restrict__ Q,
                                                   const unsigned short* __restrict__ Kb,
                                                   const unsigned short* __restrict__ Vb,
                                                   const float* __restrict__ qr,
                                                   const float* __restrict__ relbias,
                                                   unsigned short* __restrict__ Obf,
                                                   int S, int Hh) {
    __shared__ unsigned short Qs[64 * 72];
    __shared__ unsigned short Ks[64 * 72];
    __shared__ unsigned short Vt[64 * 72];       // transposed: [d][k]
    __shared__ unsigned short Pb[64 * 72];       // probabilities, bf16
    __shared__ float Sc[64 * SCC];               // scores
    __shared__ float rowMax[64], rowSum[64], rowFac[64];
    __shared__ float rbias[33];

    const int tid  = threadIdx.x;
    const int lane = tid & 31, half = lane >> 4, l16 = lane & 15;
    const int w = tid >> 5, wr = w >> 1, wc = w & 1;

    const int qtiles = S / 64;
    const int bh = blockIdx.x / qtiles;
    const int q0 = (blockIdx.x % qtiles) * 64;
    const int b = bh / Hh, hh = bh % Hh;

    if (tid < 33) rbias[tid] = relbias[tid];
    if (tid < 64) { rowMax[tid] = -1e30f; rowSum[tid] = 0.f; }

    const int srow = tid >> 2, sc0 = (tid & 3) * 16;  // 64x64 bf16 tile stager

    // stage Q tile (async)
    {
        const unsigned short* src = Q + ((size_t)((b * S + q0 + srow) * Hh + hh)) * 64 + sc0;
        cp_b128(src,     &Qs[srow * 72 + sc0]);
        cp_b128(src + 8, &Qs[srow * 72 + sc0 + 8]);
    }
    wait_cp();
    __syncthreads();

    // persistent Q fragments for this wave's q-row subtile (kk = 0, 32)
    v16bf qf[2];
#pragma unroll
    for (int kk = 0; kk < 2; ++kk)
        qf[kk] = ldfrag(&Qs[(wr * 16 + l16) * 72 + kk * 32], half);

    v8f oacc[2] = {};

    for (int kt = 0; kt < S; kt += 64) {
        // stage K tile (async), V tile (transposed in VGPRs)
        {
            const unsigned short* ks = Kb + ((size_t)((b * S + kt + srow) * Hh + hh)) * 64 + sc0;
            cp_b128(ks,     &Ks[srow * 72 + sc0]);
            cp_b128(ks + 8, &Ks[srow * 72 + sc0 + 8]);
            const unsigned short* vs = Vb + ((size_t)((b * S + kt + srow) * Hh + hh)) * 64 + sc0;
            unsigned short tmp[16];
            *(uint4*)&tmp[0] = ((const uint4*)vs)[0];
            *(uint4*)&tmp[8] = ((const uint4*)vs)[1];
#pragma unroll
            for (int i = 0; i < 16; ++i) Vt[(sc0 + i) * 72 + srow] = tmp[i];
        }
        wait_cp();
        __syncthreads();

        // scores: wave w computes 16x16 tiles (rt=wr, ct in {2wc, 2wc+1})
#pragma unroll
        for (int c = 0; c < 2; ++c) {
            int ct = wc * 2 + c;
            v8f sacc = {};
#pragma unroll
            for (int kk = 0; kk < 2; ++kk) {
                v16bf kf = ldfrag(&Ks[(ct * 16 + l16) * 72 + kk * 32], half);
                sacc = wmma_bf16(qf[kk], kf, sacc);
            }
#pragma unroll
            for (int r = 0; r < 8; ++r)
                Sc[(wr * 16 + r + 8 * half) * SCC + ct * 16 + l16] = sacc[r];
        }
        __syncthreads();

        // online softmax: one thread per q-row; P emitted directly as bf16
        if (tid < 64) {
            const int row  = tid;
            const int qabs = q0 + row;
            const float* qrp = qr + ((size_t)((b * S + qabs) * Hh + hh)) * 36;
            float mprev = rowMax[row];
            float tmax  = mprev;
            for (int c = 0; c < 64; ++c) {
                int dd = (kt + c) - qabs;
                if (dd > 16) dd = 16;
                if (dd < -16) dd = -16;
                dd += 16;
                float s = (Sc[row * SCC + c] + qrp[dd] + rbias[dd]) * 0.125f;
                Sc[row * SCC + c] = s;
                tmax = fmaxf(tmax, s);
            }
            float fac = __expf(mprev - tmax);     // mprev=-1e30 first pass -> 0
            float sum = rowSum[row] * fac;
            for (int c = 0; c < 64; ++c) {
                float p = __expf(Sc[row * SCC + c] - tmax);
                Pb[row * 72 + c] = f2bf(p);
                sum += p;
            }
            rowMax[row] = tmax; rowSum[row] = sum; rowFac[row] = fac;
        }
        __syncthreads();

        // rescale O, then O += P @ V   (wave owns rows wr*16.., cols wc*32..)
#pragma unroll
        for (int j = 0; j < 2; ++j)
#pragma unroll
            for (int r = 0; r < 8; ++r)
                oacc[j][r] *= rowFac[wr * 16 + r + 8 * half];
#pragma unroll
        for (int kk = 0; kk < 2; ++kk) {
            v16bf pf = ldfrag(&Pb[(wr * 16 + l16) * 72 + kk * 32], half);
#pragma unroll
            for (int j = 0; j < 2; ++j) {
                v16bf vf = ldfrag(&Vt[(wc * 32 + j * 16 + l16) * 72 + kk * 32], half);
                oacc[j] = wmma_bf16(pf, vf, oacc[j]);
            }
        }
        __syncthreads();
    }

    // epilogue: normalize and store bf16 [B,S,H,DH]
#pragma unroll
    for (int j = 0; j < 2; ++j) {
        int dcol = wc * 32 + j * 16 + l16;
#pragma unroll
        for (int r = 0; r < 8; ++r) {
            int row = wr * 16 + r + 8 * half;
            float v = oacc[j][r] / rowSum[row];
            Obf[((size_t)((b * S + q0 + row) * Hh + hh)) * 64 + dcol] = f2bf(v);
        }
    }
}

// ---------------------------------------------------------------------------
// Host orchestration
// ---------------------------------------------------------------------------
extern "C" void kernel_launch(void* const* d_in, const int* in_sizes, int n_in,
                              void* d_out, int out_size, void* d_ws, size_t ws_size,
                              hipStream_t stream) {
    (void)in_sizes; (void)n_in; (void)out_size; (void)ws_size;
    constexpr int B = 4, S = 1024, D = 1024, H = 16, MLP = 4096;
    constexpr int MN = B * S;                       // 4096 rows

    const float* x    = (const float*)d_in[0];
    const float* ln1s = (const float*)d_in[1];
    const float* ln1b = (const float*)d_in[2];
    const float* wq   = (const float*)d_in[3];
    const float* bq   = (const float*)d_in[4];
    const float* wk   = (const float*)d_in[5];
    const float* bk   = (const float*)d_in[6];
    const float* wv   = (const float*)d_in[7];
    const float* bv   = (const float*)d_in[8];
    const float* wo   = (const float*)d_in[9];
    const float* bo   = (const float*)d_in[10];
    const float* rel  = (const float*)d_in[11];
    const float* relb = (const float*)d_in[12];
    const float* ln2s = (const float*)d_in[13];
    const float* ln2b = (const float*)d_in[14];
    const float* w1   = (const float*)d_in[15];
    const float* b1   = (const float*)d_in[16];
    const float* w2   = (const float*)d_in[17];
    const float* b2   = (const float*)d_in[18];

    char* ws = (char*)d_ws;
    size_t off = 0;
    auto alloc = [&](size_t bytes) -> void* {
        void* p = ws + off;
        off = (off + bytes + 255) & ~(size_t)255;
        return p;
    };
    unsigned short* h_bf  = (unsigned short*)alloc((size_t)MN * D * 2);
    unsigned short* wq_bf = (unsigned short*)alloc((size_t)D * D * 2);
    unsigned short* wk_bf = (unsigned short*)alloc((size_t)D * D * 2);
    unsigned short* wv_bf = (unsigned short*)alloc((size_t)D * D * 2);
    unsigned short* wo_bf = (unsigned short*)alloc((size_t)D * D * 2);
    unsigned short* w1_bf = (unsigned short*)alloc((size_t)D * MLP * 2);
    unsigned short* w2_bf = (unsigned short*)alloc((size_t)MLP * D * 2);
    unsigned short* q_bf  = (unsigned short*)alloc((size_t)MN * D * 2);
    unsigned short* k_bf  = (unsigned short*)alloc((size_t)MN * D * 2);
    unsigned short* v_bf  = (unsigned short*)alloc((size_t)MN * D * 2);
    float*          qr_f  = (float*)alloc((size_t)MN * H * 36 * 4);
    unsigned short* at_bf = (unsigned short*)alloc((size_t)MN * D * 2);
    float*          x1    = (float*)alloc((size_t)MN * D * 4);
    unsigned short* y_bf  = (unsigned short*)alloc((size_t)MN * D * 2);
    unsigned short* m1_bf = (unsigned short*)alloc((size_t)MN * MLP * 2);

    // weights -> bf16
    k_f32_to_bf16<<<(D * D + 255) / 256, 256, 0, stream>>>(wq, wq_bf, D * D);
    k_f32_to_bf16<<<(D * D + 255) / 256, 256, 0, stream>>>(wk, wk_bf, D * D);
    k_f32_to_bf16<<<(D * D + 255) / 256, 256, 0, stream>>>(wv, wv_bf, D * D);
    k_f32_to_bf16<<<(D * D + 255) / 256, 256, 0, stream>>>(wo, wo_bf, D * D);
    k_f32_to_bf16<<<(D * MLP + 255) / 256, 256, 0, stream>>>(w1, w1_bf, D * MLP);
    k_f32_to_bf16<<<(MLP * D + 255) / 256, 256, 0, stream>>>(w2, w2_bf, MLP * D);

    // LN1 -> h (bf16)
    k_layernorm_bf16<<<MN, 256, 0, stream>>>(x, ln1s, ln1b, h_bf, D);

    // QKV projections
    dim3 gDD(MN / 128, D / 64);
    k_gemm_bf16<<<gDD, 256, 0, stream>>>(h_bf, wq_bf, bq, nullptr, q_bf, MN, D, D, 1, 0);
    k_gemm_bf16<<<gDD, 256, 0, stream>>>(h_bf, wk_bf, bk, nullptr, k_bf, MN, D, D, 1, 0);
    k_gemm_bf16<<<gDD, 256, 0, stream>>>(h_bf, wv_bf, bv, nullptr, v_bf, MN, D, D, 1, 0);

    // relative-position dot table
    const int qrows = MN * H;
    k_qr<<<((size_t)qrows * 33 + 255) / 256, 256, 0, stream>>>(q_bf, rel, qr_f, qrows);

    // flash attention
    k_attention<<<B * H * (S / 64), 256, 0, stream>>>(q_bf, k_bf, v_bf, qr_f, relb, at_bf, S, H);

    // output projection + residual -> x1 (fp32)
    k_gemm_bf16<<<gDD, 256, 0, stream>>>(at_bf, wo_bf, bo, x, x1, MN, D, D, 0, 0);

    // LN2 -> y (bf16)
    k_layernorm_bf16<<<MN, 256, 0, stream>>>(x1, ln2s, ln2b, y_bf, D);

    // MLP
    dim3 gDM(MN / 128, MLP / 64);
    k_gemm_bf16<<<gDM, 256, 0, stream>>>(y_bf, w1_bf, b1, nullptr, m1_bf, MN, MLP, D, 1, 1);
    k_gemm_bf16<<<gDD, 256, 0, stream>>>(m1_bf, w2_bf, b2, x1, (float*)d_out, MN, D, MLP, 0, 0);
}
